// SimplicialClosedRoPE_47193100649131
// MI455X (gfx1250) — matrix operand order repaired and checked by reading, not compiled, not run on hardware
//
#include <hip/hip_runtime.h>

typedef __attribute__((ext_vector_type(16))) _Float16 v16h;
typedef __attribute__((ext_vector_type(8)))  float    v8f;

namespace {
constexpr int kB = 2, kT = 160, kH = 8, kF = 16;
constexpr int kN = kB * kH * kT * kF;   // 40960 phase entries per table
constexpr int kTiles = kT / 16;         // 10 tiles of 16 along each T axis
}

// ---------------------------------------------------------------------------
// Kernel 1: phase tables.  th_mu = pos . mu, th_nu = pos . nu, then store
// cos/sin of both into workspace: [cmu | smu | cnu | snu], each kN floats.
// 41K threads, trivially memory bound; runs once so the main kernel does
// only FMAs (cos(a-b) = ca*cb + sa*sb) instead of per-pair transcendentals.
// ---------------------------------------------------------------------------
__global__ __launch_bounds__(256) void build_phase_tables(
    const float* __restrict__ pos, const float* __restrict__ mu,
    const float* __restrict__ nu, float* __restrict__ ws) {
  int idx = blockIdx.x * blockDim.x + threadIdx.x;
  if (idx >= kN) return;
  int f = idx % kF;
  int t = (idx / kF) % kT;
  int h = (idx / (kF * kT)) % kH;
  int b = idx / (kF * kT * kH);
  const float* p = pos + ((size_t)b * kT + t) * 3;
  const float* m = mu  + ((size_t)h * kF + f) * 3;
  const float* n = nu  + ((size_t)h * kF + f) * 3;
  float tm = fmaf(p[0], m[0], fmaf(p[1], m[1], p[2] * m[2]));
  float tn = fmaf(p[0], n[0], fmaf(p[1], n[1], p[2] * n[2]));
  float cm, sm, cn, sn;
  __sincosf(tm, &sm, &cm);
  __sincosf(tn, &sn, &cn);
  ws[idx]            = cm;
  ws[idx + kN]       = sm;
  ws[idx + 2 * kN]   = cn;
  ws[idx + 3 * kN]   = sn;
}

// ---------------------------------------------------------------------------
// Kernel 2: one block per (b,h,q).  residual[j,k] = left(160x32) @ right^T.
// Rank 32 == K of v_wmma_f32_16x16x32_f16 -> one WMMA per 16x16 output tile.
// Wave w (of 10) owns M-tile w: A fragment built once, 10 B fragments + WMMAs.
// Fragments are synthesized in registers straight from LDS-resident cos/sin
// tables following the CDNA5 16-bit A / B VGPR layouts (CDNA5 ISA):
//   A 16x32: lanes 0-15 hold K=0..7,16..23; lanes 16-31 hold K=8..15,24..31
//   B 32x16: lanes 0-15 hold K=0..7,16..23; lanes 16-31 hold K=8..15,24..31
//   C/D    : VGPR i = row (i + 8*half), col = lane%16
// Output (262 MB > 192 MB L2) is written with non-temporal stores.
// ---------------------------------------------------------------------------
__global__ __launch_bounds__(320) void simplicial_rope_wmma(
    const float* __restrict__ ws, const float* __restrict__ gate_logit,
    const float* __restrict__ logit_scale, float* __restrict__ out) {
  const float* cmu = ws;
  const float* smu = ws + kN;
  const float* cnu = ws + 2 * kN;
  const float* snu = ws + 3 * kN;

  const int bhq = blockIdx.x;           // (b*H + h)*T + q
  const int q   = bhq % kT;
  const int bh  = bhq / kT;
  const int h   = bh % kH;

  __shared__ float s_cmu[kT * kF];
  __shared__ float s_smu[kT * kF];
  __shared__ float s_cnu[kT * kF];
  __shared__ float s_snu[kT * kF];
  const size_t base = (size_t)bh * kT * kF;
  for (int i = threadIdx.x; i < kT * kF; i += blockDim.x) {
    s_cmu[i] = cmu[base + i];
    s_smu[i] = smu[base + i];
    s_cnu[i] = cnu[base + i];
    s_snu[i] = snu[base + i];
  }
  __syncthreads();

  const int lane = threadIdx.x & 31;
  const int wave = threadIdx.x >> 5;    // 0..9 = M tile index
  const int half = lane >> 4;           // K-half selector per ISA layout
  const int lo   = lane & 15;

  // Query-side phases. For A (mu) each lane only ever touches f in
  // [8*half, 8*half+8); B (nu) uses the same f, though all 16 are loaded.
  float cqm[8], sqm[8];
#pragma unroll
  for (int i = 0; i < 8; ++i) {
    cqm[i] = s_cmu[q * kF + half * 8 + i];
    sqm[i] = s_smu[q * kF + half * 8 + i];
  }
  float cqn[16], sqn[16];
#pragma unroll
  for (int f = 0; f < 16; ++f) {
    cqn[f] = s_cnu[q * kF + f];
    sqn[f] = s_snu[q * kF + f];
  }

  // A fragment: rows j = wave*16 + (lane%16), K = rank index r.
  // r < 16: cos(dmu) = cq*cj + sq*sj ;  r >= 16: sin(dmu) = sq*cj - cq*sj
  // VGPR v, elem e -> r = (v&3)*2 + e + half*8 + (v>>2)*16  (v<4 <=> r<16)
  const int j = wave * 16 + lo;
  v16h afrag;
#pragma unroll
  for (int v = 0; v < 8; ++v) {
#pragma unroll
    for (int e = 0; e < 2; ++e) {
      const int fi = (v & 3) * 2 + e;         // f - 8*half
      const int f  = fi + half * 8;
      const float cj = s_cmu[j * kF + f];
      const float sj = s_smu[j * kF + f];
      const float val = (v < 4) ? fmaf(cqm[fi], cj, sqm[fi] * sj)
                                : (sqm[fi] * cj - cqm[fi] * sj);
      afrag[2 * v + e] = (_Float16)val;
    }
  }

  const float gate  = 1.0f / (1.0f + __expf(-gate_logit[h]));
  const float scale = __expf(logit_scale[h]) * gate * 0.17677669529663687f; // 32^-0.5

  float* outp = out + (size_t)bhq * kT * kT;

#pragma unroll
  for (int nt = 0; nt < kTiles; ++nt) {
    // B fragment: B[r][n] = right[t = nt*16+n][r]
    // elems 0..7 cover r=8*half+0..7    -> cos(dnu) =  cq*ct + sq*st
    // elems 8..15 cover r=16+8*half+0..7 -> -sin(dnu) = cq*st - sq*ct
    const int t = nt * 16 + lo;
    v16h bfrag;
#pragma unroll
    for (int v = 0; v < 8; ++v) {
#pragma unroll
      for (int e = 0; e < 2; ++e) {
        const int f = (v & 3) * 2 + e + half * 8;
        const float ct = s_cnu[t * kF + f];
        const float st = s_snu[t * kF + f];
        const float val = (v < 4) ? fmaf(cqn[f], ct, sqn[f] * st)
                                  : fmaf(cqn[f], st, -sqn[f] * ct);
        bfrag[2 * v + e] = (_Float16)val;
      }
    }
    v8f acc = {};
    acc = __builtin_amdgcn_wmma_f32_16x16x32_f16(
        /*neg_a=*/false, afrag, /*neg_b=*/false, bfrag,
        /*c_mod=*/(short)0, acc, /*reuse_a=*/false, /*reuse_b=*/false);
#pragma unroll
    for (int i = 0; i < 8; ++i) {
      const int row = wave * 16 + half * 8 + i;  // C/D layout
      __builtin_nontemporal_store(acc[i] * scale,
                                  outp + (size_t)row * kT + nt * 16 + lo);
    }
  }
}

extern "C" void kernel_launch(void* const* d_in, const int* in_sizes, int n_in,
                              void* d_out, int out_size, void* d_ws, size_t ws_size,
                              hipStream_t stream) {
  const float* pos         = (const float*)d_in[0];
  const float* mu          = (const float*)d_in[1];
  const float* nu          = (const float*)d_in[2];
  const float* gate_logit  = (const float*)d_in[3];
  const float* logit_scale = (const float*)d_in[4];
  float* out = (float*)d_out;
  float* ws  = (float*)d_ws;  // uses 4*kN*4 = 655,360 bytes

  build_phase_tables<<<(kN + 255) / 256, 256, 0, stream>>>(pos, mu, nu, ws);
  simplicial_rope_wmma<<<kB * kH * kT, 320, 0, stream>>>(ws, gate_logit,
                                                         logit_scale, out);
}
